// PointNet_19954418057391
// MI455X (gfx1250) — matrix operand order repaired and checked by reading, not compiled, not run on hardware
//
#include <hip/hip_runtime.h>
#include <hip/hip_bf16.h>

#define D_IN  128
#define D_HID 512
#define D_OUT 256

typedef __attribute__((ext_vector_type(16))) __bf16 v16bf;
typedef __attribute__((ext_vector_type(8)))  float  v8f;

__device__ __forceinline__ unsigned short f32_bf16(float f) {
  union { float f; unsigned u; } x; x.f = f;
  return (unsigned short)((x.u + 0x7FFFu + ((x.u >> 16) & 1u)) >> 16);  // RNE
}
__device__ __forceinline__ unsigned pack2(float a, float b) {
  return (unsigned)f32_bf16(a) | ((unsigned)f32_bf16(b) << 16);
}

// ---------------------------------------------------------------------------
// Pack W = (A - B) [512 x K] f32 into bf16 WMMA B-fragment-linear layout:
// dword index g = ((ct*KS + ks)*32 + lane)*8 + i  holds  {W[cc][k], W[cc][k+1]}
// with cc = ct*16 + (lane&15), k = ks*32 + 16*(lane>>4) + 2*i   (contiguous k order, matching the A-fragment reads)
// ---------------------------------------------------------------------------
__global__ __launch_bounds__(256) void prep_w_kernel(
    const float* __restrict__ A, const float* __restrict__ B,
    unsigned* __restrict__ Wfrag, int K) {
  int g = blockIdx.x * 256 + threadIdx.x;
  int total = 256 * K;                      // = 512*K/2 packed dwords
  if (g >= total) return;
  int i    = g & 7;
  int l    = (g >> 3) & 31;
  int rest = g >> 8;
  int KS = K / 32;
  int ks = rest % KS;
  int ct = rest / KS;
  int cc = ct * 16 + (l & 15);
  int k  = ks * 32 + ((l >> 4) << 4) + 2 * i;
  size_t idx = (size_t)cc * K + k;
  Wfrag[g] = pack2(A[idx] - B[idx], A[idx + 1] - B[idx + 1]);
}

__global__ __launch_bounds__(256) void zero_kernel(unsigned* __restrict__ p, int n) {
  int i = blockIdx.x * 256 + threadIdx.x;
  if (i < n) p[i] = 0u;
}

// Column-sum of f32 input [nrows x 128] -> per-WG partials [nwg][128]
__global__ __launch_bounds__(256) void colsum_f32_kernel(
    const float* __restrict__ X, float* __restrict__ part, int nrows) {
  __shared__ float ps[2][D_IN];
  int tid  = threadIdx.x;
  int c    = tid & (D_IN - 1);
  int half = tid >> 7;
  int rend = blockIdx.x * 256 + 256; if (rend > nrows) rend = nrows;
  float s = 0.f;
  for (int r = blockIdx.x * 256 + half; r < rend; r += 2)
    s += X[(size_t)r * D_IN + c];
  ps[half][c] = s;
  __syncthreads();
  if (tid < D_IN) part[(size_t)blockIdx.x * D_IN + tid] = ps[0][tid] + ps[1][tid];
}

// Deterministic tree reduce: [nwg][C] -> [8][C] -> [C]
__global__ __launch_bounds__(256) void reduce1_kernel(
    const float* __restrict__ part, float* __restrict__ part2, int nwg, int C) {
  int t = blockIdx.x * 256 + threadIdx.x;
  if (t >= 8 * C) return;
  int chunk = t / C, c = t % C;
  float s = 0.f;
  for (int i = chunk; i < nwg; i += 8) s += part[(size_t)i * C + c];
  part2[t] = s;
}
__global__ __launch_bounds__(256) void reduce2_kernel(
    const float* __restrict__ part2, float* __restrict__ s, int C) {
  int c = blockIdx.x * 256 + threadIdx.x;
  if (c >= C) return;
  float a = 0.f;
  for (int i = 0; i < 8; ++i) a += part2[i * C + c];
  s[c] = a;
}

// c[j] = sum_k s[k] * Bw[j][k]   (Bw is [512 x K] row-major)
__global__ __launch_bounds__(256) void bias_kernel(
    const float* __restrict__ s, const float* __restrict__ Bw,
    float* __restrict__ cvec, int K) {
  int j = blockIdx.x * 256 + threadIdx.x;
  if (j >= D_HID) return;
  float acc = 0.f;
  const float* row = Bw + (size_t)j * K;
  for (int k = 0; k < K; ++k) acc += s[k] * row[k];
  cvec[j] = acc;
}

// out[o] = sum_c max_bits(c) * ow[o][c] + ob[o]
__global__ __launch_bounds__(256) void out_kernel(
    const unsigned* __restrict__ hmaxbits, const float* __restrict__ ow,
    const float* __restrict__ ob, float* __restrict__ out) {
  int o = threadIdx.x;
  float acc = ob[o];
  const float* row = ow + (size_t)o * D_HID;
  for (int c = 0; c < D_HID; ++c)
    acc += __uint_as_float(hmaxbits[c]) * row[c];
  out[o] = acc;
}

// ---------------------------------------------------------------------------
// Fused layer: Y = X @ W^T + c ; LayerNorm ; ReLU ; (store bf16 + colsum) or
// (column max via bitwise atomicMax, valid since y >= 0).
// WG = 256 threads = 8 waves; tile = 64 rows x 512 cols; wave w owns cols
// [w*64, w*64+64) across all 4 row-strips -> acc[4][4] of v8f.
// ---------------------------------------------------------------------------
template <int K, bool IN_F32, bool DO_MAX>
__global__ __launch_bounds__(256) void layer_kernel(
    const void* __restrict__ Xin, const uint4* __restrict__ Wfrag,
    const float* __restrict__ cvec, const float* __restrict__ gvec,
    const float* __restrict__ bvec, unsigned short* __restrict__ Hout,
    float* __restrict__ colpart, unsigned* __restrict__ hmax, int nrows) {
  constexpr int P  = K + 8;   // LDS pitch (bf16 elems): keeps b128 align, spreads banks
  constexpr int KS = K / 32;
  __shared__ __align__(16) unsigned short Xs[64 * P];
  __shared__ float wpart[8][64][2];
  __shared__ float rowstat[64][2];

  const int tid  = threadIdx.x;
  const int row0 = blockIdx.x * 64;
  const int lane = tid & 31;
  const int w    = tid >> 5;
  const int lhalf = lane >> 4;
  const int l15   = lane & 15;

  // ---- stage X tile (64 x K) into LDS as bf16, zero-pad invalid rows ----
  if (IN_F32) {
    const float4* Xg = (const float4*)Xin;
    constexpr int CPR = K / 4;
    for (int i = tid; i < 64 * CPR; i += 256) {
      int r = i / CPR, cc = i % CPR;
      int gr = row0 + r;
      float4 v = (gr < nrows) ? Xg[(size_t)gr * CPR + cc] : make_float4(0.f, 0.f, 0.f, 0.f);
      unsigned* dst = (unsigned*)&Xs[r * P + cc * 4];
      dst[0] = pack2(v.x, v.y);
      dst[1] = pack2(v.z, v.w);
    }
  } else {
    const uint4* Xg = (const uint4*)Xin;  // 8 bf16 per uint4
    constexpr int CPR = K / 8;
    for (int i = tid; i < 64 * CPR; i += 256) {
      int r = i / CPR, cc = i % CPR;
      int gr = row0 + r;
      uint4 v = (gr < nrows) ? Xg[(size_t)gr * CPR + cc] : make_uint4(0u, 0u, 0u, 0u);
      *(uint4*)&Xs[r * P + cc * 8] = v;
    }
  }
  __syncthreads();

  v8f acc[4][4];
  v8f zero = {0.f, 0.f, 0.f, 0.f, 0.f, 0.f, 0.f, 0.f};
#pragma unroll
  for (int rs = 0; rs < 4; ++rs)
#pragma unroll
    for (int j = 0; j < 4; ++j) acc[rs][j] = zero;

  // ---- K loop: 4 A-frags (LDS) + 4 B-frags (global, fragment-linear) -> 16 WMMA
  for (int ks = 0; ks < KS; ++ks) {
    v16bf afr[4];
    const int kb = ks * 32 + 16 * lhalf;  // contiguous k: lane half at +16, matches W frag layout
#pragma unroll
    for (int rs = 0; rs < 4; ++rs) {
      int r = rs * 16 + l15;
      union { uint4 q[2]; v16bf v; } ua;
      ua.q[0] = *(const uint4*)&Xs[r * P + kb];
      ua.q[1] = *(const uint4*)&Xs[r * P + kb + 8];
      afr[rs] = ua.v;
    }
#pragma unroll
    for (int j = 0; j < 4; ++j) {
      int ct = w * 4 + j;
      size_t base = ((size_t)(ct * KS + ks) * 32 + lane) * 2;  // in uint4 units
      union { uint4 q[2]; v16bf v; } ub;
      ub.q[0] = Wfrag[base];
      ub.q[1] = Wfrag[base + 1];
      v16bf bfr = ub.v;
#pragma unroll
      for (int rs = 0; rs < 4; ++rs)
        acc[rs][j] = __builtin_amdgcn_wmma_f32_16x16x32_bf16(
            false, afr[rs], false, bfr, (short)0, acc[rs][j], false, false);
    }
  }

  // ---- epilogue: + c, LayerNorm stats (shfl + LDS cross-wave), transform ----
  float cv[4], gv[4], bv[4];
#pragma unroll
  for (int j = 0; j < 4; ++j) {
    int col = (w * 4 + j) * 16 + l15;
    cv[j] = cvec[col]; gv[j] = gvec[col]; bv[j] = bvec[col];
  }
#pragma unroll
  for (int rs = 0; rs < 4; ++rs)
#pragma unroll
    for (int j = 0; j < 4; ++j)
#pragma unroll
      for (int v = 0; v < 8; ++v) acc[rs][j][v] += cv[j];

#pragma unroll
  for (int rs = 0; rs < 4; ++rs) {
#pragma unroll
    for (int v = 0; v < 8; ++v) {
      float x0 = acc[rs][0][v], x1 = acc[rs][1][v], x2 = acc[rs][2][v], x3 = acc[rs][3][v];
      float sm = x0 + x1 + x2 + x3;
      float sq = x0 * x0 + x1 * x1 + x2 * x2 + x3 * x3;
      for (int d = 1; d < 16; d <<= 1) {
        sm += __shfl_xor(sm, d, 32);
        sq += __shfl_xor(sq, d, 32);
      }
      if (l15 == 0) {
        int r = rs * 16 + v + 8 * lhalf;
        wpart[w][r][0] = sm;
        wpart[w][r][1] = sq;
      }
    }
  }
  __syncthreads();
  if (tid < 64) {
    float sm = 0.f, sq = 0.f;
    for (int ww = 0; ww < 8; ++ww) { sm += wpart[ww][tid][0]; sq += wpart[ww][tid][1]; }
    float mu  = sm * (1.f / (float)D_HID);
    float var = sq * (1.f / (float)D_HID) - mu * mu;
    rowstat[tid][0] = mu;
    rowstat[tid][1] = rsqrtf(var + 1e-5f);
  }
  __syncthreads();

  float csum[4] = {0.f, 0.f, 0.f, 0.f};
  float cmax[4] = {0.f, 0.f, 0.f, 0.f};
#pragma unroll
  for (int rs = 0; rs < 4; ++rs) {
#pragma unroll
    for (int v = 0; v < 8; ++v) {
      int r = rs * 16 + v + 8 * lhalf;
      float mu = rowstat[r][0], rsig = rowstat[r][1];
      bool valid = (row0 + r) < nrows;
#pragma unroll
      for (int j = 0; j < 4; ++j) {
        float y = (acc[rs][j][v] - mu) * rsig * gv[j] + bv[j];
        y = fmaxf(y, 0.f);
        if (DO_MAX) {
          cmax[j] = fmaxf(cmax[j], valid ? y : 0.f);
        } else {
          if (valid) {
            int col = (w * 4 + j) * 16 + l15;
            Hout[(size_t)(row0 + r) * D_HID + col] = f32_bf16(y);
          }
          csum[j] += valid ? y : 0.f;
        }
      }
    }
  }
  // lanes l and l+16 hold the same column (different rows): fold halves.
#pragma unroll
  for (int j = 0; j < 4; ++j) {
    int col = (w * 4 + j) * 16 + l15;
    if (DO_MAX) {
      float m = fmaxf(cmax[j], __shfl_xor(cmax[j], 16, 32));
      if (lhalf == 0) atomicMax(&hmax[col], __float_as_uint(m));  // y>=0 -> monotone bits
    } else {
      float s = csum[j] + __shfl_xor(csum[j], 16, 32);
      if (lhalf == 0) colpart[(size_t)blockIdx.x * D_HID + col] = s;
    }
  }
}

// ---------------------------------------------------------------------------
extern "C" void kernel_launch(void* const* d_in, const int* in_sizes, int n_in,
                              void* d_out, int out_size, void* d_ws, size_t ws_size,
                              hipStream_t stream) {
  const float* in_set = (const float*)d_in[0];
  const float* A0 = (const float*)d_in[1];
  const float* B0 = (const float*)d_in[2];
  const float* A1 = (const float*)d_in[3];
  const float* B1 = (const float*)d_in[4];
  const float* A2 = (const float*)d_in[5];
  const float* B2 = (const float*)d_in[6];
  const float* gam = (const float*)d_in[7];
  const float* bet = (const float*)d_in[8];
  const float* ow  = (const float*)d_in[9];
  const float* ob  = (const float*)d_in[10];
  float* out = (float*)d_out;

  const int n    = in_sizes[0] / D_IN;       // 100000
  const int nwg  = (n + 63) / 64;            // layer GEMM workgroups
  const int nwg0 = (n + 255) / 256;          // input colsum workgroups

  char* ws = (char*)d_ws;
  size_t off = 0;
  auto take = [&](size_t bytes) -> char* {
    char* p = ws + off;
    off += (bytes + 255) & ~(size_t)255;
    return p;
  };
  unsigned*       W0f   = (unsigned*)take((size_t)256 * D_IN  * 4);
  unsigned*       W1f   = (unsigned*)take((size_t)256 * D_HID * 4);
  unsigned*       W2f   = (unsigned*)take((size_t)256 * D_HID * 4);
  float*          cvec  = (float*)take(D_HID * 4);
  float*          svec  = (float*)take(D_HID * 4);
  unsigned*       hmax  = (unsigned*)take(D_HID * 4);
  float*          part  = (float*)take((size_t)nwg * D_HID * 4);
  float*          part2 = (float*)take(8 * D_HID * 4);
  unsigned short* h1    = (unsigned short*)take((size_t)n * D_HID * 2);
  unsigned short* h2    = (unsigned short*)take((size_t)n * D_HID * 2);

  // weight prep (bf16 fragment-linear) + init max accumulator
  prep_w_kernel<<<(256 * D_IN  + 255) / 256, 256, 0, stream>>>(A0, B0, W0f, D_IN);
  prep_w_kernel<<<(256 * D_HID + 255) / 256, 256, 0, stream>>>(A1, B1, W1f, D_HID);
  prep_w_kernel<<<(256 * D_HID + 255) / 256, 256, 0, stream>>>(A2, B2, W2f, D_HID);
  zero_kernel<<<2, 256, 0, stream>>>(hmax, D_HID);

  // layer 0: s0 = colsum(in_set); c0 = s0 @ B0^T; GEMM+LN+ReLU -> h1 (+colsums)
  colsum_f32_kernel<<<nwg0, 256, 0, stream>>>(in_set, part, n);
  reduce1_kernel<<<(8 * D_IN + 255) / 256, 256, 0, stream>>>(part, part2, nwg0, D_IN);
  reduce2_kernel<<<1, 256, 0, stream>>>(part2, svec, D_IN);
  bias_kernel<<<2, 256, 0, stream>>>(svec, B0, cvec, D_IN);
  layer_kernel<D_IN, true, false><<<nwg, 256, 0, stream>>>(
      in_set, (const uint4*)W0f, cvec, gam, bet, h1, part, nullptr, n);

  // layer 1
  reduce1_kernel<<<(8 * D_HID + 255) / 256, 256, 0, stream>>>(part, part2, nwg, D_HID);
  reduce2_kernel<<<2, 256, 0, stream>>>(part2, svec, D_HID);
  bias_kernel<<<2, 256, 0, stream>>>(svec, B1, cvec, D_HID);
  layer_kernel<D_HID, false, false><<<nwg, 256, 0, stream>>>(
      h1, (const uint4*)W1f, cvec, gam, bet, h2, part, nullptr, n);

  // layer 2 (no store; fused column max)
  reduce1_kernel<<<(8 * D_HID + 255) / 256, 256, 0, stream>>>(part, part2, nwg, D_HID);
  reduce2_kernel<<<2, 256, 0, stream>>>(part2, svec, D_HID);
  bias_kernel<<<2, 256, 0, stream>>>(svec, B2, cvec, D_HID);
  layer_kernel<D_HID, false, true><<<nwg, 256, 0, stream>>>(
      h2, (const uint4*)W2f, cvec, gam, bet, nullptr, nullptr, hmax, n);

  // output projection
  out_kernel<<<1, 256, 0, stream>>>(hmax, ow, ob, out);
}
